// metapath_specific_64476049047836
// MI455X (gfx1250) — matrix operand order repaired and checked by reading, not compiled, not run on hardware
//
#include <hip/hip_runtime.h>
#include <hip/hip_bf16.h>
#include <math.h>

typedef __bf16 bf16_t;
typedef bf16_t v16bf __attribute__((ext_vector_type(16)));
typedef bf16_t v4bf  __attribute__((ext_vector_type(4)));
typedef float  v8f   __attribute__((ext_vector_type(8)));

#define OUT_DIM   64
#define NUM_HEADS 8
#define HID       512
#define LSEQ      4
#define ALPHA     0.01f
#define MTILE     64           // edges per block
#define HS        (HID + 32)   // LDS row stride (bf16 elems), keeps 16B alignment

#if __has_builtin(__builtin_amdgcn_tanhf)
#define TANHF(x) __builtin_amdgcn_tanhf(x)   // gfx1250 V_TANH_F32 (TRANS pipe)
#else
#define TANHF(x) tanhf(x)
#endif

union FragBF { v16bf v; uint4 q[2]; };

// ---------------- K0a: weight conversion + fused bias ----------------
__global__ void k_prep(const float* __restrict__ W_ih, const float* __restrict__ W_hh,
                       const float* __restrict__ b_ih, const float* __restrict__ b_hh,
                       bf16_t* __restrict__ wih16, bf16_t* __restrict__ whh16,
                       float* __restrict__ bc) {
    int i = blockIdx.x * blockDim.x + threadIdx.x;
    if (i < HID * HID)     whh16[i] = (bf16_t)W_hh[i];
    if (i < HID * OUT_DIM) wih16[i] = (bf16_t)W_ih[i];
    if (i < HID)           bc[i]    = b_ih[i] + b_hh[i];
}

// ---------------- K0b: reset output + softmax state every call ----------------
__global__ void k_init(float* __restrict__ out, float* __restrict__ node_max,
                       float* __restrict__ denom, int n_out, int n_nh) {
    int i = blockIdx.x * blockDim.x + threadIdx.x;
    if (i < n_out) out[i] = 0.0f;
    if (i < n_nh)  { node_max[i] = -INFINITY; denom[i] = 0.0f; }
}

// ---------------- K1: WMMA tanh-RNN over L=4 steps ----------------
// Block: 512 threads = 16 waves. Tile: 64 edges x 512 hidden.
// Wave layout: rowgroup = wave>>2 (16 rows each), colchunk = wave&3 (128 cols each).
// Each wave owns a 16x128 output slab = 8 WMMA 16x16 f32 accumulators.
__global__ __launch_bounds__(512)
void k_rnn(const float* __restrict__ features, const int* __restrict__ indices,
           const bf16_t* __restrict__ wih16, const bf16_t* __restrict__ whh16,
           const float* __restrict__ bc, float* __restrict__ hidden, int E) {
    __shared__ __align__(16) bf16_t Xsh[LSEQ][MTILE][OUT_DIM]; // gathered inputs, all steps
    __shared__ __align__(16) bf16_t Hsh[MTILE][HS];            // bf16 hidden state

    const int e0  = blockIdx.x * MTILE;
    const int tid = threadIdx.x;

    // Cooperative gather: float4 loads, packed 4xbf16 LDS stores.
    for (int idx = tid; idx < MTILE * LSEQ * (OUT_DIM / 4); idx += 512) {
        int i   = idx >> 6;         // edge within tile (0..63)
        int rem = idx & 63;
        int t   = rem >> 4;         // step
        int d4  = rem & 15;         // float4 group
        int e   = e0 + i;
        float4 f = make_float4(0.f, 0.f, 0.f, 0.f);
        if (e < E)
            f = *(const float4*)&features[(size_t)indices[e * LSEQ + t] * OUT_DIM + d4 * 4];
        v4bf pk;
        pk[0] = (bf16_t)f.x; pk[1] = (bf16_t)f.y; pk[2] = (bf16_t)f.z; pk[3] = (bf16_t)f.w;
        *(v4bf*)&Xsh[t][i][d4 * 4] = pk;
    }
    __syncthreads();

    const int lane = tid & 31;
    const int wave = tid >> 5;      // 0..15
    const int half = lane >> 4;     // lane group
    const int mr   = lane & 15;     // A: row / B,C,D: column
    const int m0   = (wave >> 2) * 16;   // row-group base (0,16,32,48)
    const int n0   = (wave & 3) * 128;   // column-chunk base (0,128,256,384)

    v8f acc[8];

    for (int step = 0; step < LSEQ; ++step) {
        #pragma unroll
        for (int t = 0; t < 8; ++t)
            #pragma unroll
            for (int r = 0; r < 8; ++r) acc[t][r] = 0.0f;

        // Z += X_t @ W_ih^T   (K = 64)
        #pragma unroll
        for (int k0 = 0; k0 < OUT_DIM; k0 += 32) {
            FragBF a;
            a.q[0] = *(const uint4*)&Xsh[step][m0 + mr][k0 + half * 8];
            a.q[1] = *(const uint4*)&Xsh[step][m0 + mr][k0 + 16 + half * 8];
            #pragma unroll
            for (int t = 0; t < 8; ++t) {
                int n = n0 + t * 16 + mr;
                FragBF b;
                const uint4* bp = (const uint4*)&wih16[n * OUT_DIM + k0 + half * 8];
                b.q[0] = bp[0]; b.q[1] = bp[2];
                acc[t] = __builtin_amdgcn_wmma_f32_16x16x32_bf16(
                    false, a.v, false, b.v, (short)0, acc[t], false, false);
            }
        }

        // Z += H_{t-1} @ W_hh^T   (K = 512); h0 == 0 so skip at step 0
        if (step > 0) {
            for (int k0 = 0; k0 < HID; k0 += 32) {
                FragBF a;
                a.q[0] = *(const uint4*)&Hsh[m0 + mr][k0 + half * 8];
                a.q[1] = *(const uint4*)&Hsh[m0 + mr][k0 + 16 + half * 8];
                #pragma unroll
                for (int t = 0; t < 8; ++t) {
                    int n = n0 + t * 16 + mr;
                    FragBF b;
                    const uint4* bp = (const uint4*)&whh16[n * HID + k0 + half * 8];
                    b.q[0] = bp[0]; b.q[1] = bp[2];
                    acc[t] = __builtin_amdgcn_wmma_f32_16x16x32_bf16(
                        false, a.v, false, b.v, (short)0, acc[t], false, false);
                }
            }
        }
        __syncthreads();   // all reads of Hsh complete before overwrite

        // h = tanh(Z + b_ih + b_hh); write bf16 back to LDS (and f32 out at last step)
        #pragma unroll
        for (int t = 0; t < 8; ++t) {
            int n = n0 + t * 16 + mr;
            float bcn = bc[n];
            #pragma unroll
            for (int r = 0; r < 8; ++r) {
                float v = TANHF(acc[t][r] + bcn);
                int m = m0 + r + half * 8;
                Hsh[m][n] = (bf16_t)v;
                if (step == LSEQ - 1 && (e0 + m) < E)
                    hidden[(size_t)(e0 + m) * HID + n] = v;
            }
        }
        __syncthreads();
    }
}

// ---------------- K2: attention logits + per-node max ----------------
__global__ void k_logits(const float* __restrict__ hidden, const float* __restrict__ attn,
                         const int* __restrict__ dst, float* __restrict__ alog,
                         float* __restrict__ node_max, int E) {
    int i = blockIdx.x * blockDim.x + threadIdx.x;
    if (i >= E * NUM_HEADS) return;
    int e = i >> 3, h = i & 7;
    const float4* hp = (const float4*)(hidden + (size_t)e * HID + h * OUT_DIM);
    const float4* ap = (const float4*)(attn + h * OUT_DIM);
    float s = 0.0f;
    #pragma unroll
    for (int q = 0; q < OUT_DIM / 4; ++q) {
        float4 a4 = ap[q];
        float4 h4 = hp[q];
        s += h4.x * a4.x + h4.y * a4.y + h4.z * a4.z + h4.w * a4.w;
    }
    s = (s >= 0.0f) ? s : ALPHA * s;          // LeakyReLU
    alog[i] = s;
    atomicMax(&node_max[dst[e] * NUM_HEADS + h], s);
}

// ---------------- K3: exp(a - max) + per-node denominator ----------------
__global__ void k_expsum(const float* __restrict__ alog, const float* __restrict__ node_max,
                         const int* __restrict__ dst, float* __restrict__ aexp,
                         float* __restrict__ denom, int E) {
    int i = blockIdx.x * blockDim.x + threadIdx.x;
    if (i >= E * NUM_HEADS) return;
    int e = i >> 3, h = i & 7;
    int nb = dst[e] * NUM_HEADS + h;
    float v = expf(alog[i] - node_max[nb]);
    aexp[i] = v;
    atomicAdd(&denom[nb], v);
}

// ---------------- K4: weighted scatter-sum into destination nodes ----------------
// One thread per (edge, 16-element group); a group never crosses a head boundary.
__global__ void k_scatter(const float* __restrict__ hidden, const float* __restrict__ aexp,
                          const float* __restrict__ denom, const int* __restrict__ dst,
                          float* __restrict__ out, int E) {
    int i = blockIdx.x * blockDim.x + threadIdx.x;
    if (i >= E * (HID / 16)) return;
    int e = i >> 5;                 // 32 groups per edge
    int g = i & 31;
    int n = g * 16;
    int h = n >> 6;
    int d2 = dst[e];
    float w = aexp[e * NUM_HEADS + h] / denom[d2 * NUM_HEADS + h];
    const float4* hp = (const float4*)(hidden + (size_t)e * HID + n);
    float* op = out + (size_t)d2 * HID + n;
    #pragma unroll
    for (int q = 0; q < 4; ++q) {
        float4 v = hp[q];
        atomicAdd(op + q * 4 + 0, v.x * w);
        atomicAdd(op + q * 4 + 1, v.y * w);
        atomicAdd(op + q * 4 + 2, v.z * w);
        atomicAdd(op + q * 4 + 3, v.w * w);
    }
}

extern "C" void kernel_launch(void* const* d_in, const int* in_sizes, int n_in,
                              void* d_out, int out_size, void* d_ws, size_t ws_size,
                              hipStream_t stream) {
    const float* features = (const float*)d_in[0];
    const float* W_ih     = (const float*)d_in[1];
    const float* W_hh     = (const float*)d_in[2];
    const float* b_ih     = (const float*)d_in[3];
    const float* b_hh     = (const float*)d_in[4];
    const float* attn     = (const float*)d_in[5];
    const int*   indices  = (const int*)d_in[6];
    const int*   dst      = (const int*)d_in[7];

    const int E = in_sizes[7];                      // number of edges
    const int N = out_size / (NUM_HEADS * OUT_DIM); // number of nodes

    // Workspace carve-up
    char*  ws     = (char*)d_ws;
    size_t off    = 0;
    auto   carve  = [&](size_t bytes) { void* p = ws + off; off = (off + bytes + 255) & ~(size_t)255; return p; };
    float*  hidden   = (float*)carve((size_t)E * HID * sizeof(float));
    bf16_t* wih16    = (bf16_t*)carve((size_t)HID * OUT_DIM * sizeof(bf16_t));
    bf16_t* whh16    = (bf16_t*)carve((size_t)HID * HID * sizeof(bf16_t));
    float*  bc       = (float*)carve(HID * sizeof(float));
    float*  alog     = (float*)carve((size_t)E * NUM_HEADS * sizeof(float));
    float*  aexp     = (float*)carve((size_t)E * NUM_HEADS * sizeof(float));
    float*  node_max = (float*)carve((size_t)N * NUM_HEADS * sizeof(float));
    float*  denom    = (float*)carve((size_t)N * NUM_HEADS * sizeof(float));

    const int T = 256;
    // K0a: weight conversion
    k_prep<<<(HID * HID + T - 1) / T, T, 0, stream>>>(W_ih, W_hh, b_ih, b_hh, wih16, whh16, bc);
    // K0b: reset output + softmax state
    {
        int n_out = N * NUM_HEADS * OUT_DIM;
        int n_nh  = N * NUM_HEADS;
        int mx = n_out > n_nh ? n_out : n_nh;
        k_init<<<(mx + T - 1) / T, T, 0, stream>>>((float*)d_out, node_max, denom, n_out, n_nh);
    }
    // K1: RNN via WMMA (512 threads per block, 64 edges per block)
    k_rnn<<<(E + MTILE - 1) / MTILE, 512, 0, stream>>>(features, indices, wih16, whh16, bc, hidden, E);
    // K2..K4: edge softmax + scatter
    k_logits<<<((size_t)E * NUM_HEADS + T - 1) / T, T, 0, stream>>>(hidden, attn, dst, alog, node_max, E);
    k_expsum<<<((size_t)E * NUM_HEADS + T - 1) / T, T, 0, stream>>>(alog, node_max, dst, aexp, denom, E);
    k_scatter<<<((size_t)E * (HID / 16) + T - 1) / T, T, 0, stream>>>(hidden, aexp, denom, dst, (float*)d_out, E);
}
